// SparseMOE_80582176408229
// MI455X (gfx1250) — matrix operand order repaired and checked by reading, not compiled, not run on hardware
//
#include <hip/hip_runtime.h>
#include <hip/hip_bf16.h>

typedef __attribute__((ext_vector_type(16))) __bf16 v16bf;
typedef __attribute__((ext_vector_type(8)))  __bf16 v8bf;
typedef __attribute__((ext_vector_type(8)))  float  v8f;

#define T_TOK 8192
#define H_DIM 1024
#define E_NUM 8

// ---------------- zero the final-output region (harness poisons d_out) ----------------
__global__ void moe_zero_kernel(float4* __restrict__ p, size_t n4) {
    size_t i = (size_t)blockIdx.x * blockDim.x + threadIdx.x;
    size_t stride = (size_t)gridDim.x * blockDim.x;
    for (; i < n4; i += stride) p[i] = make_float4(0.f, 0.f, 0.f, 0.f);
}

// ---------------- x: f32 -> bf16 (row-major preserved) ----------------
__global__ void moe_cvt_x_kernel(const float* __restrict__ x, __bf16* __restrict__ xb, size_t n) {
    size_t i = ((size_t)blockIdx.x * blockDim.x + threadIdx.x) * 4;
    if (i < n) {
        float4 v = *(const float4*)(x + i);
        xb[i + 0] = (__bf16)v.x;
        xb[i + 1] = (__bf16)v.y;
        xb[i + 2] = (__bf16)v.z;
        xb[i + 3] = (__bf16)v.w;
    }
}

// ---------------- expert_w [E][d][f] f32 -> [E][f][d] bf16 (LDS tile transpose) ------
__global__ void moe_xpose_w_kernel(const float* __restrict__ w, __bf16* __restrict__ wt) {
    __shared__ float tile[32][33];
    int e  = blockIdx.z;
    int d0 = blockIdx.y * 32;
    int f0 = blockIdx.x * 32;
    int tx = threadIdx.x;      // 0..31
    int ty = threadIdx.y;      // 0..7
    const float* src = w + ((size_t)e * H_DIM + d0) * H_DIM + f0;
#pragma unroll
    for (int i = 0; i < 4; ++i)
        tile[ty + 8 * i][tx] = src[(size_t)(ty + 8 * i) * H_DIM + tx];
    __syncthreads();
    __bf16* dst = wt + ((size_t)e * H_DIM + f0) * H_DIM + d0;
#pragma unroll
    for (int i = 0; i < 4; ++i)
        dst[(size_t)(ty + 8 * i) * H_DIM + tx] = (__bf16)tile[tx][ty + 8 * i];
}

// ---------------- router: logits (raw -> d_out), softmax top-2, normalized weights ----
__global__ void moe_router_kernel(const float* __restrict__ x, const float* __restrict__ gw,
                                  const float* __restrict__ gb, float* __restrict__ out_logits,
                                  int* __restrict__ route_e, float* __restrict__ route_w) {
    __shared__ float s_gw[H_DIM * E_NUM];   // 32 KB
    int tid = threadIdx.x;
    for (int i = tid; i < H_DIM * E_NUM; i += 256) s_gw[i] = gw[i];
    __syncthreads();

    int t = blockIdx.x * 256 + tid;
    float acc[E_NUM];
#pragma unroll
    for (int e = 0; e < E_NUM; ++e) acc[e] = gb[e];

    const float* xr = x + (size_t)t * H_DIM;
    for (int k = 0; k < H_DIM; k += 4) {
        float4 xv = *(const float4*)(xr + k);
#pragma unroll
        for (int e = 0; e < E_NUM; ++e) {
            acc[e] += xv.x * s_gw[(k + 0) * E_NUM + e]
                    + xv.y * s_gw[(k + 1) * E_NUM + e]
                    + xv.z * s_gw[(k + 2) * E_NUM + e]
                    + xv.w * s_gw[(k + 3) * E_NUM + e];
        }
    }
#pragma unroll
    for (int e = 0; e < E_NUM; ++e) out_logits[(size_t)t * E_NUM + e] = acc[e];

    // softmax (shift for stability), then top-2; normalization cancels in the ratio
    float mx = acc[0];
#pragma unroll
    for (int e = 1; e < E_NUM; ++e) mx = fmaxf(mx, acc[e]);
    float p[E_NUM];
#pragma unroll
    for (int e = 0; e < E_NUM; ++e) p[e] = __expf(acc[e] - mx);

    int e0 = 0;
#pragma unroll
    for (int e = 1; e < E_NUM; ++e) if (p[e] > p[e0]) e0 = e;
    int e1 = (e0 == 0) ? 1 : 0;
#pragma unroll
    for (int e = 0; e < E_NUM; ++e) if (e != e0 && p[e] > p[e1]) e1 = e;

    float ws = p[e0] + p[e1];
    route_e[2 * t]     = e0;
    route_e[2 * t + 1] = e1;
    route_w[2 * t]     = p[e0] / ws;
    route_w[2 * t + 1] = p[e1] / ws;
}

// ---------------- deterministic per-expert compaction (token-order prefix scan) -------
// Pads each expert's token list to a multiple of 32 (GEMM M-tile granularity).
__global__ void moe_compact_kernel(const int* __restrict__ route_e, const float* __restrict__ route_w,
                                   int* __restrict__ list, float* __restrict__ wlist,
                                   int* __restrict__ cntp) {
    int e   = blockIdx.x;
    int tid = threadIdx.x;
    __shared__ int s_scan[256];
    __shared__ int s_base;
    if (tid == 0) s_base = 0;
    __syncthreads();

    for (int t0 = 0; t0 < T_TOK; t0 += 256) {
        int t = t0 + tid;
        int flag = 0;
        float w = 0.f;
        int a = route_e[2 * t], b = route_e[2 * t + 1];
        if (a == e)      { flag = 1; w = route_w[2 * t]; }
        else if (b == e) { flag = 1; w = route_w[2 * t + 1]; }

        s_scan[tid] = flag;
        __syncthreads();
        for (int off = 1; off < 256; off <<= 1) {     // inclusive Hillis-Steele scan
            int add = (tid >= off) ? s_scan[tid - off] : 0;
            __syncthreads();
            s_scan[tid] += add;
            __syncthreads();
        }
        if (flag) {
            int pos = s_base + s_scan[tid] - 1;
            list[e * T_TOK + pos]  = t;
            wlist[e * T_TOK + pos] = w;
        }
        __syncthreads();
        if (tid == 0) s_base += s_scan[255];
        __syncthreads();
    }
    int cnt    = s_base;
    int padded = (cnt + 31) & ~31;
    if (tid < padded - cnt) {                          // dummy rows: token 0, weight 0
        list[e * T_TOK + cnt + tid]  = 0;
        wlist[e * T_TOK + cnt + tid] = 0.f;
    }
    if (tid == 0) cntp[e] = padded;
}

// ---------------- expert GEMM: gathered tokens, bf16 WMMA, 32Mx32N register tile ------
// Block = 128 threads (4 waves). Each wave owns a 32Mx32N register tile:
//   2 A frags x 2 B frags -> 4 WMMAs per K-step, 8 b128 loads per K-step
//   (2 loads per WMMA; A and B fragments each reused twice).
// Block covers 32M x 128N. Grid: (H/128, T/32 worst case, E); early-exit past count.
__global__ void moe_gemm_kernel(const __bf16* __restrict__ xb, const __bf16* __restrict__ wtb,
                                const float* __restrict__ eb, const int* __restrict__ list,
                                const float* __restrict__ wlist, const int* __restrict__ cntp,
                                float* __restrict__ out) {
    int e  = blockIdx.z;
    int m0 = blockIdx.y * 32;
    if (m0 >= cntp[e]) return;

    int wave = threadIdx.x >> 5;
    int lane = threadIdx.x & 31;
    int hi   = lane >> 4;          // half-wave: selects K sub-block
    int l15  = lane & 15;

    int nb  = blockIdx.x * 128 + wave * 32;        // wave's N base
    int n0c = nb + l15;                            // B col for frag b0
    int n1c = nb + 16 + l15;                       // B col for frag b1

    int tok0 = list[e * T_TOK + m0 + l15];         // A row for frag a0
    int tok1 = list[e * T_TOK + m0 + 16 + l15];    // A row for frag a1

    // A fragment (16-bit A 16x32): lane holds its row; K chunks [kb..kb+7] and
    // [kb+16..kb+23] with kb = hi*8.
    const __bf16* arow0 = xb + (size_t)tok0 * H_DIM + hi * 8;
    const __bf16* arow1 = xb + (size_t)tok1 * H_DIM + hi * 8;
    // B fragment (16-bit B 32x16): lane holds its col; K chunks [kb..kb+7] and
    // [kb+16..kb+23] with kb = hi*8.
    const __bf16* brow0 = wtb + ((size_t)e * H_DIM + n0c) * H_DIM + hi * 8;
    const __bf16* brow1 = wtb + ((size_t)e * H_DIM + n1c) * H_DIM + hi * 8;

    v8f c00 = {}, c01 = {}, c10 = {}, c11 = {};
    for (int kk = 0; kk < H_DIM; kk += 32) {
        union { v16bf v; v8bf h[2]; } a0, a1, b0, b1;
        a0.h[0] = *(const v8bf*)(arow0 + kk);
        a0.h[1] = *(const v8bf*)(arow0 + kk + 16);
        a1.h[0] = *(const v8bf*)(arow1 + kk);
        a1.h[1] = *(const v8bf*)(arow1 + kk + 16);
        b0.h[0] = *(const v8bf*)(brow0 + kk);
        b0.h[1] = *(const v8bf*)(brow0 + kk + 16);
        b1.h[0] = *(const v8bf*)(brow1 + kk);
        b1.h[1] = *(const v8bf*)(brow1 + kk + 16);
        c00 = __builtin_amdgcn_wmma_f32_16x16x32_bf16(false, a0.v, false, b0.v, (short)0, c00, false, false);
        c01 = __builtin_amdgcn_wmma_f32_16x16x32_bf16(false, a0.v, false, b1.v, (short)0, c01, false, false);
        c10 = __builtin_amdgcn_wmma_f32_16x16x32_bf16(false, a1.v, false, b0.v, (short)0, c10, false, false);
        c11 = __builtin_amdgcn_wmma_f32_16x16x32_bf16(false, a1.v, false, b1.v, (short)0, c11, false, false);
    }

    float bias0 = eb[e * H_DIM + n0c];
    float bias1 = eb[e * H_DIM + n1c];
    int rbase   = hi * 8;          // D layout: VGPR r -> row r (lanes 0-15) / row 8+r (lanes 16-31)

#pragma unroll
    for (int r = 0; r < 8; ++r) {
        // M sub-tile 0 (rows m0 .. m0+15)
        int mm   = m0 + rbase + r;
        int tk   = list[e * T_TOK + mm];
        float wg = wlist[e * T_TOK + mm];
        atomicAdd(out + (size_t)tk * H_DIM + n0c, wg * (c00[r] + bias0));
        atomicAdd(out + (size_t)tk * H_DIM + n1c, wg * (c01[r] + bias1));
        // M sub-tile 1 (rows m0+16 .. m0+31)
        int mm2   = m0 + 16 + rbase + r;
        int tk2   = list[e * T_TOK + mm2];
        float wg2 = wlist[e * T_TOK + mm2];
        atomicAdd(out + (size_t)tk2 * H_DIM + n0c, wg2 * (c10[r] + bias0));
        atomicAdd(out + (size_t)tk2 * H_DIM + n1c, wg2 * (c11[r] + bias1));
    }
}

extern "C" void kernel_launch(void* const* d_in, const int* in_sizes, int n_in,
                              void* d_out, int out_size, void* d_ws, size_t ws_size,
                              hipStream_t stream) {
    const float* x  = (const float*)d_in[0];   // [4,2048,1024]
    const float* gw = (const float*)d_in[1];   // [1024,8]
    const float* gb = (const float*)d_in[2];   // [8]
    const float* ew = (const float*)d_in[3];   // [8,1024,1024]
    const float* eb = (const float*)d_in[4];   // [8,1024]

    float* out_final  = (float*)d_out;                       // [T,H]
    float* out_logits = out_final + (size_t)T_TOK * H_DIM;   // [T,E]

    char* ws = (char*)d_ws;
    size_t off = 0;
    __bf16* xb  = (__bf16*)(ws + off); off += (size_t)T_TOK * H_DIM * 2;           // 16 MB
    __bf16* wtb = (__bf16*)(ws + off); off += (size_t)E_NUM * H_DIM * H_DIM * 2;   // 16 MB
    int*   route_e = (int*)  (ws + off); off += (size_t)2 * T_TOK * 4;
    float* route_w = (float*)(ws + off); off += (size_t)2 * T_TOK * 4;
    int*   list    = (int*)  (ws + off); off += (size_t)E_NUM * T_TOK * 4;
    float* wlist   = (float*)(ws + off); off += (size_t)E_NUM * T_TOK * 4;
    int*   cntp    = (int*)  (ws + off);

    moe_zero_kernel<<<4096, 256, 0, stream>>>((float4*)out_final, (size_t)T_TOK * H_DIM / 4);
    moe_cvt_x_kernel<<<(T_TOK * H_DIM / 4) / 256, 256, 0, stream>>>(x, xb, (size_t)T_TOK * H_DIM);
    moe_xpose_w_kernel<<<dim3(H_DIM / 32, H_DIM / 32, E_NUM), dim3(32, 8), 0, stream>>>(ew, wtb);
    moe_router_kernel<<<T_TOK / 256, 256, 0, stream>>>(x, gw, gb, out_logits, route_e, route_w);
    moe_compact_kernel<<<E_NUM, 256, 0, stream>>>(route_e, route_w, list, wlist, cntp);
    moe_gemm_kernel<<<dim3(H_DIM / 128, T_TOK / 32, E_NUM), 128, 0, stream>>>(
        xb, wtb, eb, list, wlist, cntp, out_final);
}
